// MultiHeadAttention_51565377356341
// MI455X (gfx1250) — matrix operand order repaired and checked by reading, not compiled, not run on hardware
//
#include <hip/hip_runtime.h>
#include <hip/hip_bf16.h>

// ---------------------------------------------------------------------------
// MI455X (gfx1250) fused multi-head attention, bf16 WMMA + f32 accumulate.
//   B=4, S=2048, D=1024, H=16, depth=64
// d_out = out[B,S,D] (8,388,608 f32) ++ attn[B,H,S,S] (268,435,456 f32)
// Workspace: Qb/Kb/Vb (bf16 inputs) + Qh/Kh/Vt/Om (bf16 intermediates)
//            7 x 16 MB + 4 x 2 MB transposed bf16 weights = ~120 MB.
// Roofline: mandatory traffic ~1.2 GB (attn write dominates) ~= 52 us at
// 23.3 TB/s; 138 GFLOP of bf16 WMMA is comparable -> keep attn resident in
// LDS between softmax and AV (never re-read from HBM), stream attn out with
// NT stores so the 192 MB L2 keeps the reused Q/K/V/Om working set.
// ---------------------------------------------------------------------------

typedef __attribute__((ext_vector_type(16))) __bf16 v16bf;
typedef __attribute__((ext_vector_type(8)))  __bf16 v8bf;
typedef __attribute__((ext_vector_type(4)))  __bf16 v4bf;
typedef __attribute__((ext_vector_type(8)))  float  v8f;
typedef __attribute__((ext_vector_type(4)))  float  v4f;

__device__ __forceinline__ __bf16 f2bf(float f) {
  union { float f; unsigned u; } in; in.f = f;
  unsigned u = in.u;
  unsigned r = u + 0x7FFFu + ((u >> 16) & 1u);   // round-to-nearest-even
  union { unsigned short s; __bf16 b; } out;
  out.s = (unsigned short)(r >> 16);
  return out.b;
}

// A-fragment chunk: 8 consecutive bf16 (16B -> one b128 load).
__device__ __forceinline__ void load_a8(const __bf16* p, v16bf& a, int off) {
  v8bf t = *(const v8bf*)p;
#pragma unroll
  for (int j = 0; j < 8; ++j) a[off + j] = t[j];
}

// B-fragment: 16 consecutive bf16 (this lane's K-run for its column).
__device__ __forceinline__ v16bf load_b16(const __bf16* p) {
  v8bf lo = *(const v8bf*)p;
  v8bf hi = *(const v8bf*)(p + 8);
  v16bf b;
#pragma unroll
  for (int j = 0; j < 8; ++j) { b[j] = lo[j]; b[8 + j] = hi[j]; }
  return b;
}

__device__ __forceinline__ v8f wmma_bf16(v16bf a, v16bf b, v8f c) {
  return __builtin_amdgcn_wmma_f32_16x16x32_bf16(
      false, a, false, b, (short)0, c, false, false);
}

// ---------------------------------------------------------------------------
// Weight transpose + f32->bf16:  Wt[n*1024+k] = bf16(W[k*1024+n])
// ---------------------------------------------------------------------------
__global__ void wcvt_kernel(const float* __restrict__ W, __bf16* __restrict__ Wt) {
  int idx = blockIdx.x * blockDim.x + threadIdx.x;   // n*1024 + k
  int n = idx >> 10, k = idx & 1023;
  Wt[idx] = f2bf(W[(k << 10) + n]);
}

// ---------------------------------------------------------------------------
// Activation f32->bf16 (one-time, so projection GEMMs never re-convert).
// 4 elements per thread; NT loads (read-once data, keep L2 for bf16 copies).
// ---------------------------------------------------------------------------
__global__ void xcvt_kernel(const float* __restrict__ X, __bf16* __restrict__ Xb) {
  int i4 = (blockIdx.x * blockDim.x + threadIdx.x) * 4;
  v4f v = __builtin_nontemporal_load((const v4f*)(X + i4));
  v4bf o;
#pragma unroll
  for (int j = 0; j < 4; ++j) o[j] = f2bf(v[j]);
  *(v4bf*)(Xb + i4) = o;
}

// ---------------------------------------------------------------------------
// Projection GEMM: out = A[8192x1024](bf16) * W[1024x1024] + bias
//   MODE 0: bf16 out, head layout   [B,H,S,64]   (Q, K)
//   MODE 1: bf16 out, transposed    [B,H,64,S]   (V)
//   MODE 2: f32 out (NT), plain [8192,1024]      (final projection)
// Block = 256 thr (8 waves); wave = 16-row strip x 64 cols; K-chunks of 32.
// A (16MB bf16) + weights (2MB bf16) are L2-resident -> fragment re-reads
// are L2 hits; HBM traffic stays compulsory-only.
// ---------------------------------------------------------------------------
template <int MODE>
__global__ void __launch_bounds__(256) proj_kernel(
    const __bf16* __restrict__ A, const __bf16* __restrict__ Wt,
    const float* __restrict__ bias, void* __restrict__ out_ptr) {
  const int lane = threadIdx.x & 31;
  const int wave = threadIdx.x >> 5;
  const int hi   = (lane >= 16);
  const int m0   = blockIdx.x * 128 + wave * 16;
  const int n0   = blockIdx.y * 64;
  const int mrow = m0 + (lane & 15);

  v8f acc[4] = {};

  for (int kc = 0; kc < 1024; kc += 32) {
    v16bf a;
    const int base0 = kc + (hi ? 16 : 0);
    const __bf16* ap = A + (size_t)mrow * 1024 + base0;
    load_a8(ap,      a, 0);    // K = base0 .. base0+7
    load_a8(ap + 8,  a, 8);    // K = base0+8 .. base0+15
    const int kb = kc + (hi ? 16 : 0);
    const __bf16* bp = Wt + (size_t)(n0 + (lane & 15)) * 1024 + kb;
#pragma unroll
    for (int t = 0; t < 4; ++t)
      acc[t] = wmma_bf16(a, load_b16(bp + (size_t)t * 16 * 1024), acc[t]);
  }

  float*  outf = (float*)out_ptr;
  __bf16* outb = (__bf16*)out_ptr;
#pragma unroll
  for (int t = 0; t < 4; ++t) {
#pragma unroll
    for (int i = 0; i < 8; ++i) {
      int r = m0 + i + (hi ? 8 : 0);
      int c = n0 + t * 16 + (lane & 15);
      float v = acc[t][i] + bias[c];
      if (MODE == 2) {
        __builtin_nontemporal_store(v, outf + (size_t)r * 1024 + c);
      } else {
        int b = r >> 11, s = r & 2047;
        int h = c >> 6,  d = c & 63;
        size_t bh = (size_t)b * 16 + h;
        size_t idx = (MODE == 0) ? ((bh * 2048 + s) * 64 + d)
                                 : ((bh * 64 + d) * 2048 + s);
        outb[idx] = f2bf(v);
      }
    }
  }
}

// ---------------------------------------------------------------------------
// Fused attention per (b, h, 16-row q tile):
//   QK^T (WMMA) -> logits LDS (16x2048 f32) -> softmax -> NT attn writeout
//   -> AV (WMMA, A-fragments = ds_load_b128 from a packed bf16 prob copy)
//   -> merged-head bf16 output.
// LDS: 128K logits + 64K bf16 probs + 8K partials + scratch = 205,952 B
// (1 WG / WGP; 8 waves keep the XDL + VALU pipes co-executing).
// ---------------------------------------------------------------------------
__global__ void __launch_bounds__(256) attn_kernel(
    const __bf16* __restrict__ Qh, const __bf16* __restrict__ Kh,
    const __bf16* __restrict__ Vt, const float* __restrict__ mask,
    float* __restrict__ attn_out, __bf16* __restrict__ Om) {
  extern __shared__ char smem_raw[];
  float*  lgt    = (float*)smem_raw;                  // [16][2048] f32
  __bf16* pbf    = (__bf16*)(smem_raw + 131072);      // [16][2048] bf16 probs
  float*  pacc   = (float*)(smem_raw + 196608);       // [2][16][64]
  float*  red    = (float*)(smem_raw + 204800);       // [16][16]
  float*  rowmax = (float*)(smem_raw + 205824);       // [16]
  float*  rowinv = (float*)(smem_raw + 205888);       // [16]

  const int tid  = threadIdx.x;
  const int lane = tid & 31;
  const int wave = tid >> 5;
  const int hi   = (lane >= 16);

  const int bh = blockIdx.x >> 7;                     // b*16 + h
  const int qt = blockIdx.x & 127;
  const int b  = bh >> 4;
  const int h  = bh & 15;
  const int q0 = qt * 16;

  // ---- logits = (Q K^T)/8 + mask*NEG, 2-chained WMMAs per 16-key tile ----
  v16bf aq0, aq1;
  {
    const __bf16* qp = Qh + ((size_t)bh * 2048 + q0 + (lane & 15)) * 64;
    int base0 = hi ? 16 : 0;
    load_a8(qp + base0,      aq0, 0);
    load_a8(qp + base0 + 8,  aq0, 8);
    load_a8(qp + base0 + 32, aq1, 0);
    load_a8(qp + base0 + 40, aq1, 8);
  }
  const int kb0 = hi ? 16 : 0;
  for (int t = 0; t < 16; ++t) {                      // 256 keys per wave
    int n0  = wave * 256 + t * 16;
    int key = n0 + (lane & 15);
    const __bf16* kp = Kh + ((size_t)bh * 2048 + key) * 64 + kb0;
    v8f acc = {};
    acc = wmma_bf16(aq0, load_b16(kp),      acc);     // depth 0..31
    acc = wmma_bf16(aq1, load_b16(kp + 32), acc);     // depth 32..63
#pragma unroll
    for (int i = 0; i < 8; ++i) {
      int m = i + (hi ? 8 : 0);
      int n = n0 + (lane & 15);
      lgt[m * 2048 + n] = acc[i] * 0.125f + mask[b * 2048 + n] * (-1e9f);
    }
  }
  __syncthreads();

  // ---- softmax: 16 threads per row ----
  const int row = tid >> 4, slot = tid & 15;
  float lmax = -3.0e38f;
  for (int j = slot; j < 2048; j += 16) lmax = fmaxf(lmax, lgt[row * 2048 + j]);
  red[row * 16 + slot] = lmax;
  __syncthreads();
  if (slot == 0) {
    float m = -3.0e38f;
#pragma unroll
    for (int k2 = 0; k2 < 16; ++k2) m = fmaxf(m, red[row * 16 + k2]);
    rowmax[row] = m;
  }
  __syncthreads();
  float rm = rowmax[row], lsum = 0.f;
  for (int j = slot; j < 2048; j += 16) {
    float e = __expf(lgt[row * 2048 + j] - rm);
    lgt[row * 2048 + j] = e;                          // f32 for attn writeout
    pbf[row * 2048 + j] = f2bf(e);                    // packed copy for WMMA A
    lsum += e;
  }
  red[row * 16 + slot] = lsum;
  __syncthreads();
  if (slot == 0) {
    float s = 0.f;
#pragma unroll
    for (int k2 = 0; k2 < 16; ++k2) s += red[row * 16 + k2];
    rowinv[row] = 1.f / s;
  }
  __syncthreads();

  // ---- streaming (NT) float4 attn writeout, normalize on the fly ----
  {
    float* ab = attn_out + ((size_t)bh * 2048 + q0) * 2048;
#pragma unroll 4
    for (int it = 0; it < 32; ++it) {
      int f4 = tid + it * 256;
      int r  = f4 >> 9;                               // 512 float4 per row
      int c  = (f4 & 511) << 2;
      float inv = rowinv[r];
      v4f v;
      v.x = lgt[r * 2048 + c]     * inv;
      v.y = lgt[r * 2048 + c + 1] * inv;
      v.z = lgt[r * 2048 + c + 2] * inv;
      v.w = lgt[r * 2048 + c + 3] * inv;
      __builtin_nontemporal_store(v, (v4f*)(ab + (size_t)r * 2048 + c));
    }
  }

  // ---- AV: waves split as (depth tile 0..3) x (key half 0..1) ----
  const int tA  = wave & 3;
  const int kh  = wave >> 2;
  const int n0v = tA * 16;
  const __bf16* vbase = Vt + ((size_t)bh * 64 + n0v + (lane & 15)) * 2048;
  v8f acc = {};
  for (int kc = kh * 1024; kc < kh * 1024 + 1024; kc += 32) {
    v16bf a;
    int base0 = kc + (hi ? 16 : 0);
    const __bf16* lp = pbf + (lane & 15) * 2048 + base0;
    load_a8(lp,      a, 0);                           // ds_load_b128
    load_a8(lp + 8,  a, 8);                           // ds_load_b128
    int kb = kc + (hi ? 16 : 0);
    acc = wmma_bf16(a, load_b16(vbase + kb), acc);
  }
#pragma unroll
  for (int i = 0; i < 8; ++i) {
    int m = i + (hi ? 8 : 0);
    int n = n0v + (lane & 15);
    pacc[(kh * 16 + m) * 64 + n] = acc[i];
  }
  __syncthreads();

  // ---- reduce two key-halves, normalize, store merged-head bf16 ----
#pragma unroll
  for (int j = 0; j < 4; ++j) {
    int e = tid + j * 256;
    int m = e >> 6, n = e & 63;
    float v = (pacc[m * 64 + n] + pacc[(16 + m) * 64 + n]) * rowinv[m];
    Om[((size_t)b * 2048 + q0 + m) * 1024 + h * 64 + n] = f2bf(v);
  }
}

// ---------------------------------------------------------------------------
extern "C" void kernel_launch(void* const* d_in, const int* in_sizes, int n_in,
                              void* d_out, int out_size, void* d_ws, size_t ws_size,
                              hipStream_t stream) {
  const float* query = (const float*)d_in[0];
  const float* key   = (const float*)d_in[1];
  const float* value = (const float*)d_in[2];
  const float* mask  = (const float*)d_in[3];
  const float* Wq = (const float*)d_in[4];
  const float* bq = (const float*)d_in[5];
  const float* Wk = (const float*)d_in[6];
  const float* bk = (const float*)d_in[7];
  const float* Wv = (const float*)d_in[8];
  const float* bv = (const float*)d_in[9];
  const float* Wo = (const float*)d_in[10];
  const float* bo = (const float*)d_in[11];

  char* w = (char*)d_ws;
  __bf16* Qb  = (__bf16*)w; w += (size_t)8192 * 1024 * 2;   // bf16 query
  __bf16* Kb  = (__bf16*)w; w += (size_t)8192 * 1024 * 2;   // bf16 key
  __bf16* Vb  = (__bf16*)w; w += (size_t)8192 * 1024 * 2;   // bf16 value
  __bf16* Qh  = (__bf16*)w; w += (size_t)8192 * 1024 * 2;   // Q [B,H,S,64]
  __bf16* Kh  = (__bf16*)w; w += (size_t)8192 * 1024 * 2;   // K [B,H,S,64]
  __bf16* Vt  = (__bf16*)w; w += (size_t)8192 * 1024 * 2;   // V [B,H,64,S]
  __bf16* Om  = (__bf16*)w; w += (size_t)8192 * 1024 * 2;   // merged heads
  __bf16* Wqt = (__bf16*)w; w += (size_t)1024 * 1024 * 2;
  __bf16* Wkt = (__bf16*)w; w += (size_t)1024 * 1024 * 2;
  __bf16* Wvt = (__bf16*)w; w += (size_t)1024 * 1024 * 2;
  __bf16* Wot = (__bf16*)w; w += (size_t)1024 * 1024 * 2;

  float* outO = (float*)d_out;
  float* outA = outO + (size_t)4 * 2048 * 1024;

  // 1) one-time conversions (weights transposed, activations straight)
  wcvt_kernel<<<4096, 256, 0, stream>>>(Wq, Wqt);
  wcvt_kernel<<<4096, 256, 0, stream>>>(Wk, Wkt);
  wcvt_kernel<<<4096, 256, 0, stream>>>(Wv, Wvt);
  wcvt_kernel<<<4096, 256, 0, stream>>>(Wo, Wot);
  xcvt_kernel<<<8192, 256, 0, stream>>>(query, Qb);
  xcvt_kernel<<<8192, 256, 0, stream>>>(key,   Kb);
  xcvt_kernel<<<8192, 256, 0, stream>>>(value, Vb);

  // 2) projections (pure-bf16 WMMA GEMMs)
  dim3 g(64, 16);
  proj_kernel<0><<<g, 256, 0, stream>>>(Qb, Wqt, bq, (void*)Qh);
  proj_kernel<0><<<g, 256, 0, stream>>>(Kb, Wkt, bk, (void*)Kh);
  proj_kernel<1><<<g, 256, 0, stream>>>(Vb, Wvt, bv, (void*)Vt);

  // 3) fused attention (dynamic LDS: 205,952 B)
  const int smem = 205952;
  hipFuncSetAttribute((const void*)attn_kernel,
                      hipFuncAttributeMaxDynamicSharedMemorySize, smem);
  attn_kernel<<<4 * 16 * 128, 256, smem, stream>>>(Qh, Kh, Vt, mask, outA, Om);

  // 4) output projection (bf16 in, NT f32 out)
  proj_kernel<2><<<g, 256, 0, stream>>>(Om, Wot, bo, (void*)outO);
}
